// RPNModule_39015482917670
// MI455X (gfx1250) — matrix operand order repaired and checked by reading, not compiled, not run on hardware
//
#include <hip/hip_runtime.h>

// ---------------------------------------------------------------------------
// Problem constants (match reference)
// ---------------------------------------------------------------------------
#define HF 200
#define WF 304
#define CC 256
#define AA 3
#define NBATCH 2
#define NP (HF * WF)          // 60800 spatial positions per image
#define NA (NP * AA)          // 182400 anchors per image
#define XPAD 306              // W + 2 (zero pad left/right)
#define HPAD 202              // H + 2
#define KTOP 1000
#define NMS_T 0.7f
#define IMGW 1216.0f
#define IMGH 800.0f
#define CLIPV 4.135166556742356f   // log(1000/16)
#define NSEL_BLOCKS ((NA + 255) / 256)   // 713

#if __has_builtin(__builtin_amdgcn_tensor_load_to_lds) && __has_builtin(__builtin_amdgcn_s_wait_tensorcnt)
#define USE_TDM 1
#else
#define USE_TDM 0
#endif

typedef __bf16 bf16_t;
typedef bf16_t v16bf __attribute__((ext_vector_type(16)));
typedef bf16_t v8bf  __attribute__((ext_vector_type(8)));
typedef float  v8f   __attribute__((ext_vector_type(8)));
typedef unsigned v4u __attribute__((ext_vector_type(4)));
typedef int      v4i __attribute__((ext_vector_type(4)));
typedef int      v8i __attribute__((ext_vector_type(8)));

__device__ __forceinline__ unsigned short f2bf(float f) {
  unsigned u = __builtin_bit_cast(unsigned, f);
  unsigned r = u + 0x7FFFu + ((u >> 16) & 1u);   // RNE truncate f32 -> bf16
  return (unsigned short)(r >> 16);
}
__device__ __forceinline__ unsigned f2key(float f) {   // monotonic f32 -> u32
  unsigned u = __builtin_bit_cast(unsigned, f);
  return u ^ ((u & 0x80000000u) ? 0xFFFFFFFFu : 0x80000000u);
}
__device__ __forceinline__ float key2f(unsigned k) {
  unsigned u = k ^ ((k & 0x80000000u) ? 0x80000000u : 0xFFFFFFFFu);
  return __builtin_bit_cast(float, u);
}
__device__ __forceinline__ v16bf cat8(v8bf lo, v8bf hi) {
  return __builtin_shufflevector(lo, hi, 0,1,2,3,4,5,6,7,8,9,10,11,12,13,14,15);
}

// ---------------------------------------------------------------------------
// 1. Pad + convert features NCHW f32 -> NHWC bf16 with 1-px zero halo
// ---------------------------------------------------------------------------
__global__ void pad_convert_kernel(const float* __restrict__ F,
                                   unsigned short* __restrict__ Xt) {
  size_t t = (size_t)blockIdx.x * 256 + threadIdx.x;
  const size_t total = (size_t)NBATCH * HPAD * XPAD * CC;
  if (t >= total) return;
  int c  = (int)(t & 255u);
  size_t r = t >> 8;
  int xp = (int)(r % XPAD); r /= XPAD;
  int yp = (int)(r % HPAD);
  int b  = (int)(r / HPAD);
  int x = xp - 1, y = yp - 1;
  float v = 0.f;
  if (x >= 0 && x < WF && y >= 0 && y < HF)
    v = F[(((size_t)b * CC + c) * HF + y) * WF + x];
  Xt[t] = f2bf(v);
}

// ---------------------------------------------------------------------------
// 2. Repack conv weights OIHW f32 -> Wr[tap][co][ci] bf16 (ci contiguous)
// ---------------------------------------------------------------------------
__global__ void repack_w_kernel(const float* __restrict__ Wsrc,
                                unsigned short* __restrict__ Wr) {
  int t = blockIdx.x * 256 + threadIdx.x;
  if (t >= 9 * CC * CC) return;
  int ci  = t & 255;
  int co  = (t >> 8) & 255;
  int tap = t >> 16;
  int dy = tap / 3, dx = tap % 3;
  float v = Wsrc[(((size_t)co * CC + ci) * 3 + dy) * 3 + dx];
  Wr[t] = f2bf(v);
}

// ---------------------------------------------------------------------------
// 3. Fused head weights Hw[16][256] bf16 + bias Hb[16]
// ---------------------------------------------------------------------------
__global__ void repack_head_kernel(const float* __restrict__ clsW,
                                   const float* __restrict__ clsB,
                                   const float* __restrict__ bbW,
                                   const float* __restrict__ bbB,
                                   unsigned short* __restrict__ Hw,
                                   float* __restrict__ Hb) {
  int t = blockIdx.x * 256 + threadIdx.x;
  if (t >= 16 * CC) return;
  int n = t >> 8, ci = t & 255;
  float v = 0.f;
  if (n < 3)       v = clsW[n * CC + ci];
  else if (n < 15) v = bbW[(n - 3) * CC + ci];
  Hw[t] = f2bf(v);
  if (t < 16) {
    float bv = 0.f;
    if (t < 3) bv = clsB[t];
    else if (t < 15) bv = bbB[t - 3];
    Hb[t] = bv;
  }
}

// ---------------------------------------------------------------------------
// TDM tile load: 2D tile (32 ci x 64 positions) of bf16 from NHWC tensor into
// LDS, with hardware padding 8 DWORDs after every 16 DWORDs -> LDS pitch 48
// bf16 per position row (matches the WMMA A-fragment staging layout).
// D# built per CDNA5 ISA ch.8 (group0: count/lds/global/type, group1: dims).
// This toolchain exposes the 6-arg builtin: (g0, g1, g2, g3, g_extra, cpol);
// groups 2/3 unused for a 2D tensor -> zeros.
// ---------------------------------------------------------------------------
#if USE_TDM
__device__ __forceinline__ void tdm_load_tile(unsigned lds_off,
                                              unsigned long long gaddr) {
  v4u g0 = { 1u,                                       // count=1, user mode
             lds_off,                                  // lds_addr (bytes)
             (unsigned)gaddr,                          // global_addr[31:0]
             ((unsigned)(gaddr >> 32) & 0x01FFFFFFu)   // global_addr[56:32]
                 | 0x80000000u };                      // type=2 ("image")
  v8i g1 = { (int)((1u << 16)      // data_size = 1 -> 2 bytes
                 | (1u << 20)      // pad_enable
                 | (3u << 22)      // pad_interval: 16 DWORDs (one 32-ci row)
                 | (7u << 25)),    // pad_amount: 8 DWORDs (pitch 32+16 bf16)
             (int)(256u << 16),    // tensor_dim0 = 256 (bits 79:48)
             (int)(4096u << 16),   // tensor_dim1 = 4096 (bits 111:80)
             (int)(32u << 16),     // tile_dim0 = 32 ci   (bits 127:112)
             64,                   // tile_dim1 = 64 positions (bits 143:128)
             256,                  // tensor_dim0_stride = 256 (bits 207:160)
             0, 0 };
  v4i z4 = { 0, 0, 0, 0 };
  v8i z8 = { 0, 0, 0, 0, 0, 0, 0, 0 };
  __builtin_amdgcn_tensor_load_to_lds(g0, g1, z4, z4, z8, 0);
}
#endif

// ---------------------------------------------------------------------------
// 4. 3x3 conv + bias + ReLU as implicit GEMM with v_wmma_f32_16x16x32_bf16.
//    M = spatial (64/block), N = Cout (128/block, 16/wave), K = 9*256.
//    TDM double-buffers A tiles into LDS while WMMAs consume the other buffer.
// ---------------------------------------------------------------------------
__global__ __launch_bounds__(256) void conv_wmma_kernel(
    const unsigned short* __restrict__ Xt, const unsigned short* __restrict__ Wr,
    const float* __restrict__ convb, unsigned short* __restrict__ T) {
  const int tid  = threadIdx.x;
  const int wv   = tid >> 5;
  const int lane = tid & 31;
  const int g    = lane >> 4;
  const int nl   = lane & 15;
  const int x0   = blockIdx.x * 64;
  const int by   = blockIdx.y;
  const int b    = by / HF, y = by % HF;
  const int co   = blockIdx.z * 128 + wv * 16 + nl;   // N index of this lane

  v8f acc[4];
  #pragma unroll
  for (int i = 0; i < 4; ++i) acc[i] = v8f{0.f,0.f,0.f,0.f,0.f,0.f,0.f,0.f};

#if USE_TDM
  __shared__ unsigned short As[2][64 * 48];   // double-buffered, pitch 48

  // K-step s -> (tap = s/8, ci chunk c0 = (s%8)*32); tile start address
  #define CONV_STEP_ADDR(s_)                                                   \
    ((unsigned long long)(uintptr_t)(Xt +                                      \
        ((size_t)(b * HPAD + (y + ((s_) >> 3) / 3)) * XPAD +                   \
         (size_t)(x0 + ((s_) >> 3) % 3)) * CC + ((s_) & 7) * 32))

  if (wv == 0) {
    tdm_load_tile((unsigned)(uintptr_t)&As[0][0], CONV_STEP_ADDR(0));
    __builtin_amdgcn_s_wait_tensorcnt(0);
  }
  __syncthreads();

  for (int s = 0; s < 72; ++s) {
    const int buf = s & 1;
    if (wv == 0 && s + 1 < 72)   // stream next tile while we compute
      tdm_load_tile((unsigned)(uintptr_t)&As[buf ^ 1][0], CONV_STEP_ADDR(s + 1));

    const int tap = s >> 3, c0 = (s & 7) * 32;
    const unsigned short* wr = Wr + (((size_t)tap * CC + co) * CC + c0 + g * 8);
    v16bf bfrag = cat8(*(const v8bf*)wr, *(const v8bf*)(wr + 16));
    #pragma unroll
    for (int mt = 0; mt < 4; ++mt) {
      const unsigned short* ap = &As[buf][(mt * 16 + nl) * 48 + g * 8];
      v16bf afrag = cat8(*(const v8bf*)ap, *(const v8bf*)(ap + 16));
      acc[mt] = __builtin_amdgcn_wmma_f32_16x16x32_bf16(
          false, afrag, false, bfrag, (short)0, acc[mt], false, false);
    }

    if (wv == 0) __builtin_amdgcn_s_wait_tensorcnt(0);
    __syncthreads();
  }
  #undef CONV_STEP_ADDR
#else
  // Fallback: synchronous VMEM->LDS staging (round-1 path)
  __shared__ unsigned short As[64 * 48];
  const int pstage = tid >> 2, seg = tid & 3;
  for (int tap = 0; tap < 9; ++tap) {
    const int dy = tap / 3, dx = tap % 3;
    const size_t rowbase = ((size_t)(b * HPAD + (y + dy)) * XPAD) * CC;
    const int col = x0 + pstage + dx;
    for (int c0 = 0; c0 < CC; c0 += 32) {
      uint4 sv = {0u, 0u, 0u, 0u};
      if (col < XPAD) {
        const unsigned short* sp = Xt + rowbase + (size_t)col * CC + c0 + seg * 8;
        sv = *(const uint4*)sp;
        if (c0 + 32 < CC) __builtin_prefetch(sp + 32, 0, 0);
      }
      *(uint4*)(&As[pstage * 48 + seg * 8]) = sv;
      __syncthreads();
      const unsigned short* wr = Wr + (((size_t)tap * CC + co) * CC + c0 + g * 8);
      v16bf bfrag = cat8(*(const v8bf*)wr, *(const v8bf*)(wr + 16));
      #pragma unroll
      for (int mt = 0; mt < 4; ++mt) {
        const unsigned short* ap = &As[(mt * 16 + nl) * 48 + g * 8];
        v16bf afrag = cat8(*(const v8bf*)ap, *(const v8bf*)(ap + 16));
        acc[mt] = __builtin_amdgcn_wmma_f32_16x16x32_bf16(
            false, afrag, false, bfrag, (short)0, acc[mt], false, false);
      }
      __syncthreads();
    }
  }
#endif

  // epilogue: bias + ReLU, convert to bf16, store NHWC
  const float cb = convb[co];
  #pragma unroll
  for (int mt = 0; mt < 4; ++mt) {
    #pragma unroll
    for (int v = 0; v < 8; ++v) {
      int x = x0 + mt * 16 + g * 8 + v;    // D layout: m = g*8 + vgpr
      if (x < WF) {
        float val = fmaxf(acc[mt][v] + cb, 0.f);
        T[((size_t)(b * NP + y * WF + x)) * CC + co] = f2bf(val);
      }
    }
  }
}

// ---------------------------------------------------------------------------
// 5. Fused cls+bbox heads: GEMM [16 pos] x [16 out] x [K=256] per wave.
// ---------------------------------------------------------------------------
__global__ __launch_bounds__(256) void head_wmma_kernel(
    const unsigned short* __restrict__ T, const unsigned short* __restrict__ Hw,
    const float* __restrict__ Hb, float* __restrict__ HO) {
  __shared__ unsigned short Hs[16 * 264];   // pitch 264 (16B multiple, pad)
  const int tid = threadIdx.x;
  {
    int n = tid >> 4, s = tid & 15;
    const uint4* src = (const uint4*)(Hw + n * 256 + s * 16);
    uint4* dst = (uint4*)(&Hs[n * 264 + s * 16]);
    dst[0] = src[0]; dst[1] = src[1];
  }
  __syncthreads();

  const int wv = tid >> 5, lane = tid & 31, g = lane >> 4, nl = lane & 15;
  const size_t p0 = (size_t)(blockIdx.x * 8 + wv) * 16;
  v8f acc = v8f{0.f,0.f,0.f,0.f,0.f,0.f,0.f,0.f};
  const unsigned short* arow = T + (p0 + nl) * CC;

  #pragma unroll
  for (int c0 = 0; c0 < CC; c0 += 32) {
    v16bf afrag = cat8(*(const v8bf*)(arow + c0 + g * 8),
                       *(const v8bf*)(arow + c0 + 16 + g * 8));
    const unsigned short* hrow = &Hs[nl * 264 + c0 + g * 8];
    v16bf bfrag = cat8(*(const v8bf*)hrow, *(const v8bf*)(hrow + 16));
    acc = __builtin_amdgcn_wmma_f32_16x16x32_bf16(
        false, afrag, false, bfrag, (short)0, acc, false, false);
  }

  const float bias = Hb[nl];
  #pragma unroll
  for (int v = 0; v < 8; ++v) {
    size_t p = p0 + g * 8 + v;
    HO[p * 16 + nl] = acc[v] + bias;
  }
}

// ---------------------------------------------------------------------------
// 6. Decode: anchors computed analytically; scores are raw logits.
// ---------------------------------------------------------------------------
__global__ void decode_kernel(const float* __restrict__ HO,
                              float* __restrict__ SC, float* __restrict__ BX) {
  int t = blockIdx.x * 256 + threadIdx.x;
  if (t >= NBATCH * NA) return;
  int b = t / NA, n = t % NA;
  int p = n / AA, a = n % AA;
  int y = p / WF, x = p % WF;
  const float* ho = HO + ((size_t)b * NP + p) * 16;
  float logit = ho[a];
  float dxv = ho[3 + a * 4 + 0], dyv = ho[3 + a * 4 + 1];
  float dwv = fminf(ho[3 + a * 4 + 2], CLIPV);
  float dhv = fminf(ho[3 + a * 4 + 3], CLIPV);
  float s = (float)(32 << a);                 // anchor w == h == size
  float cx = x * 4.0f + 0.5f, cy = y * 4.0f + 0.5f;
  float pcx = dxv * s + cx, pcy = dyv * s + cy;
  float pw = expf(dwv) * s, ph = expf(dhv) * s;
  float* bx = BX + (size_t)t * 4;
  bx[0] = pcx - 0.5f * pw; bx[1] = pcy - 0.5f * ph;
  bx[2] = pcx + 0.5f * pw - 1.f; bx[3] = pcy + 0.5f * ph - 1.f;
  SC[t] = logit;
}

// ---------------------------------------------------------------------------
// 7. Deterministic radix top-K threshold (two 16-bit passes per image)
// ---------------------------------------------------------------------------
__global__ void hist1_kernel(const float* __restrict__ SC, unsigned* __restrict__ hist) {
  int t = blockIdx.x * 256 + threadIdx.x;
  if (t >= NBATCH * NA) return;
  int b = t / NA;
  unsigned key = f2key(SC[t]);
  atomicAdd(&hist[(size_t)b * 65536 + (key >> 16)], 1u);
}

__global__ void select1_kernel(const unsigned* __restrict__ hist, unsigned* __restrict__ ctl) {
  int b = blockIdx.x;
  const unsigned* h = hist + (size_t)b * 65536;
  __shared__ unsigned cs[256];
  unsigned ssum = 0;
  for (int i = 0; i < 256; ++i) ssum += h[threadIdx.x * 256 + i];
  cs[threadIdx.x] = ssum;
  __syncthreads();
  if (threadIdx.x == 0) {
    unsigned acc = 0; int c;
    for (c = 255; c >= 0; --c) { if (acc + cs[c] >= (unsigned)KTOP) break; acc += cs[c]; }
    if (c < 0) c = 0;
    int bin = 0; unsigned above = 0;
    for (int i = 255; i >= 0; --i) {
      unsigned hv = h[c * 256 + i];
      if (acc + hv >= (unsigned)KTOP) { bin = c * 256 + i; above = acc; break; }
      acc += hv;
    }
    ctl[b * 16 + 0] = (unsigned)bin;
    ctl[b * 16 + 1] = above;
  }
}

__global__ void hist2_kernel(const float* __restrict__ SC, const unsigned* __restrict__ ctl,
                             unsigned* __restrict__ hist) {
  int t = blockIdx.x * 256 + threadIdx.x;
  if (t >= NBATCH * NA) return;
  int b = t / NA;
  unsigned key = f2key(SC[t]);
  if ((key >> 16) == ctl[b * 16 + 0])
    atomicAdd(&hist[(size_t)b * 65536 + (key & 0xFFFFu)], 1u);
}

__global__ void select2_kernel(const unsigned* __restrict__ hist, unsigned* __restrict__ ctl) {
  int b = blockIdx.x;
  const unsigned* h = hist + (size_t)b * 65536;
  __shared__ unsigned cs[256];
  unsigned ssum = 0;
  for (int i = 0; i < 256; ++i) ssum += h[threadIdx.x * 256 + i];
  cs[threadIdx.x] = ssum;
  __syncthreads();
  if (threadIdx.x == 0) {
    unsigned above = ctl[b * 16 + 1];
    unsigned K = ((unsigned)KTOP > above) ? ((unsigned)KTOP - above) : 1u;
    unsigned acc = 0; int c;
    for (c = 255; c >= 0; --c) { if (acc + cs[c] >= K) break; acc += cs[c]; }
    if (c < 0) c = 0;
    int bin = 0;
    for (int i = 255; i >= 0; --i) {
      unsigned hv = h[c * 256 + i];
      if (acc + hv >= K) { bin = c * 256 + i; break; }
      acc += hv;
    }
    ctl[b * 16 + 2] = (ctl[b * 16 + 0] << 16) | (unsigned)bin;  // exact threshold
  }
}

// ---------------------------------------------------------------------------
// 8. Deterministic compaction: per-block counts -> serial scan -> scatter
// ---------------------------------------------------------------------------
__global__ void blockcount_kernel(const float* __restrict__ SC, const unsigned* __restrict__ ctl,
                                  unsigned* __restrict__ bc) {
  __shared__ unsigned r[256];
  int blk = blockIdx.x, b = blockIdx.y, t = threadIdx.x;
  int i = blk * 256 + t;
  unsigned thr = ctl[b * 16 + 2];
  unsigned pred = 0;
  if (i < NA) pred = (f2key(SC[(size_t)b * NA + i]) >= thr) ? 1u : 0u;
  r[t] = pred;
  __syncthreads();
  for (int s = 128; s > 0; s >>= 1) { if (t < s) r[t] += r[t + s]; __syncthreads(); }
  if (t == 0) bc[b * 1024 + blk] = r[0];
}

__global__ void scanblocks_kernel(unsigned* __restrict__ bc) {
  int b = blockIdx.x;
  unsigned acc = 0;
  for (int i = 0; i < NSEL_BLOCKS; ++i) {
    unsigned c = bc[b * 1024 + i];
    bc[b * 1024 + i] = acc;
    acc += c;
  }
}

__global__ void scatter_kernel(const float* __restrict__ SC, const unsigned* __restrict__ ctl,
                               const unsigned* __restrict__ bc,
                               unsigned long long* __restrict__ cand) {
  __shared__ unsigned pr[256];
  int blk = blockIdx.x, b = blockIdx.y, t = threadIdx.x;
  int i = blk * 256 + t;
  unsigned thr = ctl[b * 16 + 2];
  unsigned key = 0, pred = 0;
  if (i < NA) { key = f2key(SC[(size_t)b * NA + i]); pred = (key >= thr) ? 1u : 0u; }
  pr[t] = pred;
  __syncthreads();
  unsigned rank = 0;
  for (int q = 0; q < t; ++q) rank += pr[q];
  if (pred) {
    unsigned pos = bc[b * 1024 + blk] + rank;
    if (pos < 1024)
      cand[b * 1024 + pos] = ((unsigned long long)key << 32) | (unsigned)(~(unsigned)i);
  }
}

// ---------------------------------------------------------------------------
// 9. Bitonic sort 1024 entries descending; composite = (key<<32)|~idx so ties
//    break toward smaller index (matching jax top_k).
// ---------------------------------------------------------------------------
__global__ __launch_bounds__(1024) void sort_kernel(unsigned long long* __restrict__ cand) {
  __shared__ unsigned long long s[1024];
  int b = blockIdx.x, t = threadIdx.x;
  s[t] = cand[b * 1024 + t];
  __syncthreads();
  for (int k = 2; k <= 1024; k <<= 1) {
    for (int j = k >> 1; j > 0; j >>= 1) {
      int p = t ^ j;
      if (p > t) {
        unsigned long long a = s[t], c = s[p];
        bool up = ((t & k) == 0);                  // descending network
        if (up ? (a < c) : (a > c)) { s[t] = c; s[p] = a; }
      }
      __syncthreads();
    }
  }
  cand[b * 1024 + t] = s[t];
}

// ---------------------------------------------------------------------------
// 10. Gather + clip to image, emit valid flags
// ---------------------------------------------------------------------------
__global__ void clip_gather_kernel(const unsigned long long* __restrict__ cand,
                                   const float* __restrict__ BX, float* __restrict__ PB,
                                   float* __restrict__ SS, unsigned* __restrict__ VF) {
  int t = blockIdx.x * 256 + threadIdx.x;
  if (t >= NBATCH * 1024) return;
  int b = t >> 10;
  unsigned long long e = cand[t];
  unsigned key = (unsigned)(e >> 32);
  unsigned idx = ~(unsigned)(e & 0xFFFFFFFFull);
  float* pb = PB + (size_t)t * 4;
  if (key == 0u || idx >= (unsigned)NA) {
    pb[0] = pb[1] = pb[2] = pb[3] = 0.f; SS[t] = 0.f; VF[t] = 0u; return;
  }
  const float* bx = BX + ((size_t)b * NA + idx) * 4;
  float x1 = fminf(fmaxf(bx[0], 0.f), IMGW - 1.f);
  float y1 = fminf(fmaxf(bx[1], 0.f), IMGH - 1.f);
  float x2 = fminf(fmaxf(bx[2], 0.f), IMGW - 1.f);
  float y2 = fminf(fmaxf(bx[3], 0.f), IMGH - 1.f);
  pb[0] = x1; pb[1] = y1; pb[2] = x2; pb[3] = y2;
  SS[t] = key2f(key);
  float ws = x2 - x1 + 1.f, hs = y2 - y1 + 1.f;
  VF[t] = (ws >= 0.f && hs >= 0.f) ? 1u : 0u;
}

// ---------------------------------------------------------------------------
// 11. NMS: 32x32 IoU bitmask tiles (wave32-sized blocks), then sequential scan
// ---------------------------------------------------------------------------
__global__ void nms_mask_kernel(const float* __restrict__ PB, unsigned* __restrict__ MASK) {
  __shared__ float sx1[32], sy1[32], sx2[32], sy2[32], sar[32];
  int cb = blockIdx.x, rb = blockIdx.y, b = blockIdx.z, t = threadIdx.x;
  int j = cb * 32 + t;
  const float* pb = PB + ((size_t)b * 1024 + j) * 4;
  float x1 = pb[0], y1 = pb[1], x2 = pb[2], y2 = pb[3];
  sx1[t] = x1; sy1[t] = y1; sx2[t] = x2; sy2[t] = y2;
  sar[t] = (x2 - x1 + 1.f) * (y2 - y1 + 1.f);
  __syncthreads();
  int i = rb * 32 + t;
  const float* pr = PB + ((size_t)b * 1024 + i) * 4;
  float rx1 = pr[0], ry1 = pr[1], rx2 = pr[2], ry2 = pr[3];
  float rar = (rx2 - rx1 + 1.f) * (ry2 - ry1 + 1.f);
  unsigned bits = 0;
  #pragma unroll
  for (int jj = 0; jj < 32; ++jj) {
    int jg = cb * 32 + jj;
    if (jg > i) {
      float ix1 = fmaxf(rx1, sx1[jj]), iy1 = fmaxf(ry1, sy1[jj]);
      float ix2 = fminf(rx2, sx2[jj]), iy2 = fminf(ry2, sy2[jj]);
      float inter = fmaxf(ix2 - ix1 + 1.f, 0.f) * fmaxf(iy2 - iy1 + 1.f, 0.f);
      float iou = inter / (rar + sar[jj] - inter);
      if (iou > NMS_T) bits |= (1u << jj);
    }
  }
  MASK[((size_t)b * 1024 + i) * 32 + cb] = bits;
}

__global__ void nms_scan_kernel(const unsigned* __restrict__ MASK,
                                const unsigned* __restrict__ VF,
                                unsigned* __restrict__ KEEP) {
  __shared__ unsigned remv[32];
  __shared__ int curk;
  int b = blockIdx.x, t = threadIdx.x;
  remv[t] = 0u;
  __syncthreads();
  for (int i = 0; i < KTOP; ++i) {
    if (t == 0) {
      int sup = (remv[i >> 5] >> (i & 31)) & 1;
      int k = (!sup && VF[b * 1024 + i]) ? 1 : 0;
      curk = k;
      KEEP[b * 1024 + i] = (unsigned)k;
    }
    __syncthreads();
    if (curk) remv[t] |= MASK[((size_t)b * 1024 + i) * 32 + t];
    __syncthreads();
  }
}

// ---------------------------------------------------------------------------
// 12. Finalize: prefix-scan keep flags, scatter kept (box,score), zero rest
// ---------------------------------------------------------------------------
__global__ __launch_bounds__(1024) void finalize_kernel(
    const float* __restrict__ PB, const float* __restrict__ SS,
    const unsigned* __restrict__ KEEP, float* __restrict__ out) {
  __shared__ unsigned s[1024];
  int b = blockIdx.x, t = threadIdx.x;
  if (t < KTOP) {
    float* o = out + ((size_t)b * KTOP + t) * 5;
    o[0] = o[1] = o[2] = o[3] = o[4] = 0.f;
  }
  unsigned k = (t < KTOP) ? KEEP[b * 1024 + t] : 0u;
  s[t] = k;
  __syncthreads();
  for (int off = 1; off < 1024; off <<= 1) {
    unsigned v = s[t];
    unsigned a = (t >= off) ? s[t - off] : 0u;
    __syncthreads();
    s[t] = v + a;
    __syncthreads();
  }
  unsigned pos = s[t] - k;
  if (k && pos < (unsigned)KTOP) {
    const float* pb = PB + ((size_t)b * 1024 + t) * 4;
    float* o = out + ((size_t)b * KTOP + pos) * 5;
    o[0] = pb[0]; o[1] = pb[1]; o[2] = pb[2]; o[3] = pb[3];
    o[4] = SS[b * 1024 + t];
  }
}

// ---------------------------------------------------------------------------
// Workspace layout
// ---------------------------------------------------------------------------
static constexpr size_t al256(size_t x) { return (x + 255) & ~(size_t)255; }
static constexpr size_t SZ_XT   = (size_t)NBATCH * HPAD * XPAD * CC * 2;
static constexpr size_t SZ_WR   = (size_t)9 * CC * CC * 2;
static constexpr size_t SZ_HW   = (size_t)16 * CC * 2;
static constexpr size_t SZ_HB   = 16 * 4;
static constexpr size_t SZ_T    = (size_t)NBATCH * NP * CC * 2;
static constexpr size_t SZ_HO   = (size_t)NBATCH * NP * 16 * 4;
static constexpr size_t SZ_SC   = (size_t)NBATCH * NA * 4;
static constexpr size_t SZ_BX   = (size_t)NBATCH * NA * 16;
static constexpr size_t SZ_HIST = (size_t)NBATCH * 65536 * 4;
static constexpr size_t SZ_CTL  = 256;
static constexpr size_t SZ_BC   = (size_t)NBATCH * 1024 * 4;
static constexpr size_t SZ_CAND = (size_t)NBATCH * 1024 * 8;
static constexpr size_t SZ_PB   = (size_t)NBATCH * 1024 * 16;
static constexpr size_t SZ_SS   = (size_t)NBATCH * 1024 * 4;
static constexpr size_t SZ_VF   = (size_t)NBATCH * 1024 * 4;
static constexpr size_t SZ_MASK = (size_t)NBATCH * 1024 * 32 * 4;
static constexpr size_t SZ_KEEP = (size_t)NBATCH * 1024 * 4;

static constexpr size_t OFF_XT    = 0;
static constexpr size_t OFF_WR    = al256(OFF_XT + SZ_XT);
static constexpr size_t OFF_HW    = al256(OFF_WR + SZ_WR);
static constexpr size_t OFF_HB    = al256(OFF_HW + SZ_HW);
static constexpr size_t OFF_T     = al256(OFF_HB + SZ_HB);
static constexpr size_t OFF_HO    = al256(OFF_T + SZ_T);
static constexpr size_t OFF_SC    = al256(OFF_HO + SZ_HO);
static constexpr size_t OFF_BX    = al256(OFF_SC + SZ_SC);
static constexpr size_t OFF_HISTA = al256(OFF_BX + SZ_BX);
static constexpr size_t OFF_HISTB = al256(OFF_HISTA + SZ_HIST);
static constexpr size_t OFF_CTL   = al256(OFF_HISTB + SZ_HIST);
static constexpr size_t OFF_BC    = al256(OFF_CTL + SZ_CTL);
static constexpr size_t OFF_CAND  = al256(OFF_BC + SZ_BC);
static constexpr size_t OFF_PB    = al256(OFF_CAND + SZ_CAND);
static constexpr size_t OFF_SS    = al256(OFF_PB + SZ_PB);
static constexpr size_t OFF_VF    = al256(OFF_SS + SZ_SS);
static constexpr size_t OFF_MASK  = al256(OFF_VF + SZ_VF);
static constexpr size_t OFF_KEEP  = al256(OFF_MASK + SZ_MASK);

extern "C" void kernel_launch(void* const* d_in, const int* in_sizes, int n_in,
                              void* d_out, int out_size, void* d_ws, size_t ws_size,
                              hipStream_t stream) {
  const float* features = (const float*)d_in[0];
  // d_in[1] = anchors (recomputed analytically on device)
  const float* conv_w = (const float*)d_in[2];
  const float* conv_b = (const float*)d_in[3];
  const float* cls_w  = (const float*)d_in[4];
  const float* cls_b  = (const float*)d_in[5];
  const float* bbox_w = (const float*)d_in[6];
  const float* bbox_b = (const float*)d_in[7];
  float* out = (float*)d_out;

  char* w = (char*)d_ws;
  unsigned short* Xt = (unsigned short*)(w + OFF_XT);
  unsigned short* Wr = (unsigned short*)(w + OFF_WR);
  unsigned short* Hw = (unsigned short*)(w + OFF_HW);
  float* Hb          = (float*)(w + OFF_HB);
  unsigned short* T  = (unsigned short*)(w + OFF_T);
  float* HO          = (float*)(w + OFF_HO);
  float* SC          = (float*)(w + OFF_SC);
  float* BX          = (float*)(w + OFF_BX);
  unsigned* histA    = (unsigned*)(w + OFF_HISTA);
  unsigned* histB    = (unsigned*)(w + OFF_HISTB);
  unsigned* ctl      = (unsigned*)(w + OFF_CTL);
  unsigned* bc       = (unsigned*)(w + OFF_BC);
  unsigned long long* cand = (unsigned long long*)(w + OFF_CAND);
  float* PB          = (float*)(w + OFF_PB);
  float* SS          = (float*)(w + OFF_SS);
  unsigned* VF       = (unsigned*)(w + OFF_VF);
  unsigned* MASK     = (unsigned*)(w + OFF_MASK);
  unsigned* KEEP     = (unsigned*)(w + OFF_KEEP);

  // zero histograms + candidate buffer (stream ops: graph-capture safe)
  (void)hipMemsetAsync(histA, 0, SZ_HIST * 2, stream);   // histA + histB contiguous
  (void)hipMemsetAsync(cand, 0, SZ_CAND, stream);

  // --- preprocessing ---
  {
    size_t n = (size_t)NBATCH * HPAD * XPAD * CC;
    pad_convert_kernel<<<dim3((unsigned)((n + 255) / 256)), dim3(256), 0, stream>>>(features, Xt);
  }
  repack_w_kernel<<<dim3((9 * CC * CC + 255) / 256), dim3(256), 0, stream>>>(conv_w, Wr);
  repack_head_kernel<<<dim3((16 * CC + 255) / 256), dim3(256), 0, stream>>>(
      cls_w, cls_b, bbox_w, bbox_b, Hw, Hb);

  // --- WMMA conv + heads ---
  conv_wmma_kernel<<<dim3(5, NBATCH * HF, 2), dim3(256), 0, stream>>>(Xt, Wr, conv_b, T);
  head_wmma_kernel<<<dim3((NBATCH * NP) / (8 * 16)), dim3(256), 0, stream>>>(T, Hw, Hb, HO);

  // --- decode ---
  decode_kernel<<<dim3((NBATCH * NA + 255) / 256), dim3(256), 0, stream>>>(HO, SC, BX);

  // --- deterministic top-1000 selection per image ---
  hist1_kernel<<<dim3((NBATCH * NA + 255) / 256), dim3(256), 0, stream>>>(SC, histA);
  select1_kernel<<<dim3(NBATCH), dim3(256), 0, stream>>>(histA, ctl);
  hist2_kernel<<<dim3((NBATCH * NA + 255) / 256), dim3(256), 0, stream>>>(SC, ctl, histB);
  select2_kernel<<<dim3(NBATCH), dim3(256), 0, stream>>>(histB, ctl);
  blockcount_kernel<<<dim3(NSEL_BLOCKS, NBATCH), dim3(256), 0, stream>>>(SC, ctl, bc);
  scanblocks_kernel<<<dim3(NBATCH), dim3(1), 0, stream>>>(bc);
  scatter_kernel<<<dim3(NSEL_BLOCKS, NBATCH), dim3(256), 0, stream>>>(SC, ctl, bc, cand);
  sort_kernel<<<dim3(NBATCH), dim3(1024), 0, stream>>>(cand);

  // --- clip + NMS + emit ---
  clip_gather_kernel<<<dim3((NBATCH * 1024) / 256), dim3(256), 0, stream>>>(cand, BX, PB, SS, VF);
  nms_mask_kernel<<<dim3(32, 32, NBATCH), dim3(32), 0, stream>>>(PB, MASK);
  nms_scan_kernel<<<dim3(NBATCH), dim3(32), 0, stream>>>(MASK, VF, KEEP);
  finalize_kernel<<<dim3(NBATCH), dim3(1024), 0, stream>>>(PB, SS, KEEP, out);
}
